// GCN_62440234549671
// MI455X (gfx1250) — matrix operand order repaired and checked by reading, not compiled, not run on hardware
//
#include <hip/hip_runtime.h>
#include <hip/hip_bf16.h>

// ---------------- problem constants (from reference) ----------------
#define N_NODES 50000
#define N_EDGES 800000
#define DIN     41
#define DIN_PAD 64
#define HDIM    192

typedef __attribute__((ext_vector_type(16))) __bf16 v16bf;
typedef __attribute__((ext_vector_type(8)))  float  v8f;

// ---------------- degree / normalization ----------------
__global__ __launch_bounds__(256) void deg_init_kernel(float* __restrict__ deg, int n) {
  int i = blockIdx.x * blockDim.x + threadIdx.x;
  if (i < n) deg[i] = 1.0f;  // self-loop contributes 1 to every node
}

__global__ __launch_bounds__(256) void deg_accum_kernel(const int* __restrict__ dst,
                                                        float* __restrict__ deg, int e) {
  int i = blockIdx.x * blockDim.x + threadIdx.x;
  if (i < e) atomicAdd(&deg[dst[i]], 1.0f);
}

__global__ __launch_bounds__(256) void deg_rsqrt_kernel(float* __restrict__ deg, int n) {
  int i = blockIdx.x * blockDim.x + threadIdx.x;
  if (i < n) deg[i] = rsqrtf(deg[i]);  // deg >= 1 always
}

// ---------------- zero-pad x: [n,41] f32 -> [n,64] f32 ----------------
__global__ __launch_bounds__(256) void pad_x_kernel(const float* __restrict__ x,
                                                    float* __restrict__ xpad, int n) {
  int idx = blockIdx.x * blockDim.x + threadIdx.x;
  if (idx >= n * DIN_PAD) return;
  int row = idx >> 6, col = idx & 63;
  xpad[idx] = (col < DIN) ? x[(size_t)row * DIN + col] : 0.0f;
}

// ---------------- pack weights into WMMA B-fragment order (bf16) ----------------
// Bp[((ct*KSTEPS + ks)*32 + lane)*16 + i] = B[ks*32 + (i>>3)*16 + (lane>>4)*8 + (i&7)][ct*16 + (lane&15)]
// so a lane's B fragment is one contiguous 32-byte load.
__global__ __launch_bounds__(256) void pack_b_kernel(const float* __restrict__ B,
                                                     __bf16* __restrict__ Bp,
                                                     int Kpad, int Kact, int M) {
  int idx = blockIdx.x * blockDim.x + threadIdx.x;
  if (idx >= M * Kpad) return;
  int ksteps = Kpad >> 5;
  int i    = idx & 15;
  int lane = (idx >> 4) & 31;
  int rest = idx >> 9;
  int ks   = rest % ksteps;
  int ct   = rest / ksteps;
  int kk   = ks * 32 + ((i >> 3) << 4) + ((lane >> 4) << 3) + (i & 7);
  int col  = ct * 16 + (lane & 15);
  Bp[idx] = (__bf16)((kk < Kact) ? B[(size_t)kk * M + col] : 0.0f);
}

// ---------------- WMMA bf16 GEMM: C[n x M] = act(A[n x K] @ B + bias) ----------------
// One wave computes 16 rows x (CT*16) cols. K, M compile-time: no guards anywhere.
// grid.x = n/16, grid.y = M/(CT*16).
template <bool RELU, int K, int M, int CT>
__global__ __launch_bounds__(32) void wmma_gemm_kernel(const float* __restrict__ A,
                                                       const __bf16* __restrict__ Bp,
                                                       const float* __restrict__ bias,
                                                       float* __restrict__ C) {
  constexpr int KSTEPS = K / 32;
  const int lane = threadIdx.x;          // 0..31
  const int half = lane >> 4;            // 0 or 1
  const int l    = lane & 15;
  const int row0 = blockIdx.x * 16;
  const int ct0  = blockIdx.y * CT;      // first 16-col tile index

  v8f acc[CT] = {};
  // A: lane l<16 holds row row0+l, K {k..k+7, k+16..k+23}; lane l+16 holds {k+8.., k+24..}
  const float* Ap = A + (size_t)(row0 + l) * K + half * 8;
  const __bf16* Bbase = Bp + ((size_t)ct0 * KSTEPS * 32 + lane) * 16;

#pragma unroll
  for (int ks = 0; ks < KSTEPS; ++ks) {
    const float* ap = Ap + ks * 32;
    float4 a0 = *(const float4*)(ap + 0);
    float4 a1 = *(const float4*)(ap + 4);
    float4 a2 = *(const float4*)(ap + 16);
    float4 a3 = *(const float4*)(ap + 20);
    v16bf a;
    a[0]  = (__bf16)a0.x; a[1]  = (__bf16)a0.y; a[2]  = (__bf16)a0.z; a[3]  = (__bf16)a0.w;
    a[4]  = (__bf16)a1.x; a[5]  = (__bf16)a1.y; a[6]  = (__bf16)a1.z; a[7]  = (__bf16)a1.w;
    a[8]  = (__bf16)a2.x; a[9]  = (__bf16)a2.y; a[10] = (__bf16)a2.z; a[11] = (__bf16)a2.w;
    a[12] = (__bf16)a3.x; a[13] = (__bf16)a3.y; a[14] = (__bf16)a3.z; a[15] = (__bf16)a3.w;
#pragma unroll
    for (int ct = 0; ct < CT; ++ct) {
      v16bf b = *(const v16bf*)(Bbase + ((size_t)ct * KSTEPS + ks) * (32 * 16));
      acc[ct] = __builtin_amdgcn_wmma_f32_16x16x32_bf16(
          false, a, false, b, (short)0, acc[ct], false, false);
    }
  }

  // store: lane<16 -> rows row0..+7, lane>=16 -> rows row0+8..+15
  const int crow = row0 + half * 8;
#pragma unroll
  for (int ct = 0; ct < CT; ++ct) {
    int ccol = (ct0 + ct) * 16 + l;
    float bv = bias ? bias[ccol] : 0.0f;
#pragma unroll
    for (int v = 0; v < 8; ++v) {
      float val = acc[ct][v] + bv;
      if (RELU) val = fmaxf(val, 0.0f);
      C[(size_t)(crow + v) * M + ccol] = val;
    }
  }
}

// ---------------- agg = bc + dis^2 * m   (self-loop term + bias) ----------------
__global__ __launch_bounds__(256) void init_agg_kernel(const float* __restrict__ m,
                                                       const float* __restrict__ dis,
                                                       const float* __restrict__ bc,
                                                       float* __restrict__ agg, int n) {
  int idx = blockIdx.x * blockDim.x + threadIdx.x;
  if (idx >= n * HDIM) return;
  int i = idx / HDIM, f = idx % HDIM;
  float di = dis[i];
  agg[idx] = bc[f] + di * di * m[idx];
}

// ---------------- edge scatter: agg[dst] += dis[src]*dis[dst] * m[src] ----------------
// One thread per float4 feature chunk per edge; gathers/atomics resolve in L2
// (m/agg are 38.4MB each, resident in the 192MB L2).
__global__ __launch_bounds__(256) void scatter_kernel(const int* __restrict__ src,
                                                      const int* __restrict__ dst,
                                                      const float* __restrict__ dis,
                                                      const float* __restrict__ m,
                                                      float* __restrict__ agg, int e) {
  const int CH = HDIM / 4;  // 48 chunks per edge
  long long t = (long long)blockIdx.x * blockDim.x + threadIdx.x;
  int eidx = (int)(t / CH);
  int c    = (int)(t % CH);
  if (eidx >= e) return;
  int s = src[eidx], d = dst[eidx];
  float coef = dis[s] * dis[d];
  const float4 mv = *reinterpret_cast<const float4*>(m + (size_t)s * HDIM + c * 4);
  float* ap = agg + (size_t)d * HDIM + c * 4;
  atomicAdd(ap + 0, coef * mv.x);
  atomicAdd(ap + 1, coef * mv.y);
  atomicAdd(ap + 2, coef * mv.z);
  atomicAdd(ap + 3, coef * mv.w);
}

// ---------------- h += relu(agg) ----------------
__global__ __launch_bounds__(256) void residual_relu_kernel(float* __restrict__ h,
                                                            const float* __restrict__ agg,
                                                            int total) {
  int idx = blockIdx.x * blockDim.x + threadIdx.x;
  if (idx < total) h[idx] += fmaxf(agg[idx], 0.0f);
}

// ---------------- final tiny heads: pos (96->2), radius (96->1), normalize ----------------
__global__ __launch_bounds__(256) void final_head_kernel(const float* __restrict__ p2,
                                                         const float* __restrict__ r1,
                                                         const float* __restrict__ Wp3,
                                                         const float* __restrict__ bp3,
                                                         const float* __restrict__ Wr2,
                                                         const float* __restrict__ br2,
                                                         float* __restrict__ out, int n) {
  int i = blockIdx.x * blockDim.x + threadIdx.x;
  if (i >= n) return;
  const float* p = p2 + (size_t)i * 96;
  const float* r = r1 + (size_t)i * 96;
  float x = bp3[0], y = bp3[1], rr = br2[0];
#pragma unroll 4
  for (int j = 0; j < 96; ++j) {
    float pv = p[j];
    x  += pv * Wp3[j * 2 + 0];
    y  += pv * Wp3[j * 2 + 1];
    rr += r[j] * Wr2[j];
  }
  float rad  = 1.0f / (1.0f + __expf(-rr));
  float norm = sqrtf(x * x + y * y + 1e-8f);
  float s = rad / norm;
  out[(size_t)i * 2 + 0] = x * s;
  out[(size_t)i * 2 + 1] = y * s;
}

// ---------------- launch ----------------
extern "C" void kernel_launch(void* const* d_in, const int* in_sizes, int n_in,
                              void* d_out, int out_size, void* d_ws, size_t ws_size,
                              hipStream_t stream) {
  const float* x    = (const float*)d_in[0];
  const int*   ei   = (const int*)d_in[1];   // edge_index flat [2,E]
  const float* W_in = (const float*)d_in[2];
  const float* b_in = (const float*)d_in[3];
  const float* Wc   = (const float*)d_in[4]; // [4,H,H]
  const float* bc   = (const float*)d_in[5]; // [4,H]
  const float* Wp1  = (const float*)d_in[6];
  const float* bp1  = (const float*)d_in[7];
  const float* Wp2  = (const float*)d_in[8];
  const float* bp2  = (const float*)d_in[9];
  const float* Wp3  = (const float*)d_in[10];
  const float* bp3  = (const float*)d_in[11];
  const float* Wr1  = (const float*)d_in[12];
  const float* br1  = (const float*)d_in[13];
  const float* Wr2  = (const float*)d_in[14];
  const float* br2  = (const float*)d_in[15];
  float* out = (float*)d_out;

  const int n = N_NODES, e = N_EDGES;
  const int* srcIdx = ei;
  const int* dstIdx = ei + e;

  // workspace: dis | h | m | agg | packed weights   (~116 MB)
  float* ws  = (float*)d_ws;
  float* dis = ws;
  float* h   = ws + n;
  float* m   = h + (size_t)n * HDIM;
  float* agg = m + (size_t)n * HDIM;
  float* xpad = m;  // aliases m: only needed for the input GEMM, before m is first written

  __bf16* pk    = (__bf16*)(agg + (size_t)n * HDIM);
  __bf16* pWin  = pk;                      // 64*192
  __bf16* pWc   = pWin + DIN_PAD * HDIM;   // 4 * 192*192
  __bf16* pWp1  = pWc  + 4 * HDIM * HDIM;  // 192*192
  __bf16* pWr1  = pWp1 + HDIM * HDIM;      // 192*96
  __bf16* pWp2  = pWr1 + HDIM * 96;        // 192*96

  dim3 b256(256);
  const int totalNH = n * HDIM;
  const long long scatterThreads = (long long)e * (HDIM / 4);
  const int scatterBlocks = (int)((scatterThreads + 255) / 256);

  // ---- pack all weights once (tiny) ----
  pack_b_kernel<<<(DIN_PAD * HDIM + 255) / 256, b256, 0, stream>>>(W_in, pWin, DIN_PAD, DIN, HDIM);
  for (int i = 0; i < 4; ++i)
    pack_b_kernel<<<(HDIM * HDIM + 255) / 256, b256, 0, stream>>>(
        Wc + (size_t)i * HDIM * HDIM, pWc + (size_t)i * HDIM * HDIM, HDIM, HDIM, HDIM);
  pack_b_kernel<<<(HDIM * HDIM + 255) / 256, b256, 0, stream>>>(Wp1, pWp1, HDIM, HDIM, HDIM);
  pack_b_kernel<<<(HDIM * 96 + 255) / 256, b256, 0, stream>>>(Wr1, pWr1, HDIM, HDIM, 96);
  pack_b_kernel<<<(HDIM * 96 + 255) / 256, b256, 0, stream>>>(Wp2, pWp2, HDIM, HDIM, 96);

  // ---- GCN symmetric normalization (dst-degree with self-loops) ----
  deg_init_kernel<<<(n + 255) / 256, b256, 0, stream>>>(dis, n);
  deg_accum_kernel<<<(e + 255) / 256, b256, 0, stream>>>(dstIdx, dis, e);
  deg_rsqrt_kernel<<<(n + 255) / 256, b256, 0, stream>>>(dis, n);

  // ---- h = x @ W_in + b_in ----
  pad_x_kernel<<<(n * DIN_PAD + 255) / 256, b256, 0, stream>>>(x, xpad, n);
  wmma_gemm_kernel<false, DIN_PAD, HDIM, 4><<<dim3(n / 16, 3), 32, 0, stream>>>(
      xpad, pWin, b_in, h);

  // ---- 4 GCN layers ----
  for (int layer = 0; layer < 4; ++layer) {
    const float* bci = bc + (size_t)layer * HDIM;
    wmma_gemm_kernel<false, HDIM, HDIM, 4><<<dim3(n / 16, 3), 32, 0, stream>>>(
        h, pWc + (size_t)layer * HDIM * HDIM, nullptr, m);
    init_agg_kernel<<<(totalNH + 255) / 256, b256, 0, stream>>>(m, dis, bci, agg, n);
    scatter_kernel<<<scatterBlocks, b256, 0, stream>>>(srcIdx, dstIdx, dis, m, agg, e);
    residual_relu_kernel<<<(totalNH + 255) / 256, b256, 0, stream>>>(h, agg, totalNH);
  }

  // ---- heads (reuse m/agg buffers) ----
  float* p1 = m;                          // [n,192]
  float* r1 = agg;                        // [n,96]
  float* p2 = agg + (size_t)n * 96;       // [n,96]
  wmma_gemm_kernel<true, HDIM, HDIM, 4><<<dim3(n / 16, 3), 32, 0, stream>>>(h,  pWp1, bp1, p1);
  wmma_gemm_kernel<true, HDIM, 96, 2><<<dim3(n / 16, 3), 32, 0, stream>>>(h,  pWr1, br1, r1);
  wmma_gemm_kernel<true, HDIM, 96, 2><<<dim3(n / 16, 3), 32, 0, stream>>>(p1, pWp2, bp2, p2);
  final_head_kernel<<<(n + 255) / 256, b256, 0, stream>>>(p2, r1, Wp3, bp3, Wr2, br2, out, n);
}
